// TrueVBRLinear_34256659153354
// MI455X (gfx1250) — matrix operand order repaired and checked by reading, not compiled, not run on hardware
//
#include <hip/hip_runtime.h>
#include <hip/hip_fp16.h>
#include <math.h>

// ---------------------------------------------------------------------------
// TrueVBRLinear on MI455X (gfx1250)
//   out[4096, 11008] = x[4096,4096] @ W_recon.T   (fp16 out)
// Memory-bound (~730 MB total @ 23.3 TB/s ≈ 31 us floor; 0.37 TFLOP is free).
//   1) x fp32 -> fp16 once (33 MB, L2-resident afterwards)
//   2) per-row 256-entry dequant LUT (kills powf in the hot path)
//   3) W codes -> fp16 weights once (the unavoidable 360 MB int32 read)
//   4) wave32 WMMA f16 GEMM, 128x128 tile, DOUBLE-BUFFERED LDS fed by
//      gfx1250 async global->LDS copies (ASYNCcnt), M-tiles fastest so W
//      strips + x stay L2-resident.
// ---------------------------------------------------------------------------

typedef _Float16 h16;
typedef __attribute__((ext_vector_type(16))) _Float16 v16h;
typedef __attribute__((ext_vector_type(8)))  _Float16 v8h;
typedef __attribute__((ext_vector_type(8)))  float    v8f;
typedef int v4i __attribute__((vector_size(16)));   // 16B chunk for async DMA

#define GLB_AS __attribute__((address_space(1)))
#define LDS_AS __attribute__((address_space(3)))
typedef GLB_AS v4i* glb_v4i_p;
typedef LDS_AS v4i* lds_v4i_p;

#define IN_DIM   4096
#define OUT_DIM  11008
#define M_DIM    4096          // B*S = 2*2048

#define BM  128
#define BN  128
#define BK  32
#define LDK 40                 // LDS row stride in halves (80B) -> conflict-free b128

// ---- gfx1250 async global->LDS support (guarded; sync fallback otherwise) --
#if defined(__has_builtin)
#  if __has_builtin(__builtin_amdgcn_global_load_async_to_lds_b128)
#    define ASYNC_LDS 1
#    if __has_builtin(__builtin_amdgcn_s_wait_asynccnt)
#      define WAIT_ASYNC(n) __builtin_amdgcn_s_wait_asynccnt(n)
#    else
#      define WAIT_ASYNC(n) asm volatile("s_wait_asynccnt %0" ::"n"(n))
#    endif
#  endif
#endif
#ifndef ASYNC_LDS
#  define ASYNC_LDS 0
#  define WAIT_ASYNC(n)
#endif

// ------------------------- kernel 0: x fp32 -> fp16 ------------------------
__global__ __launch_bounds__(256)
void cvt_x_kernel(const float* __restrict__ x, h16* __restrict__ xh) {
    int i = (blockIdx.x * 256 + threadIdx.x) * 8;
    const float4* p = (const float4*)(x + i);
    float4 a = p[0], b = p[1];
    v8h o;
    o[0] = (h16)a.x; o[1] = (h16)a.y; o[2] = (h16)a.z; o[3] = (h16)a.w;
    o[4] = (h16)b.x; o[5] = (h16)b.y; o[6] = (h16)b.z; o[7] = (h16)b.w;
    *(v8h*)(xh + i) = o;
}

// --------------------- kernel 1: per-row dequant LUT -----------------------
__global__ __launch_bounds__(256)
void lut_kernel(const float* __restrict__ scales, const float* __restrict__ alphas,
                const float* __restrict__ powers, const float* __restrict__ divisors,
                h16* __restrict__ lut) {
    int o = blockIdx.x;
    int m = threadIdx.x;
    float xn = (float)m / divisors[o];
    float a  = alphas[o];
    float v  = (a * xn + (1.0f - a) * powf(xn, powers[o])) * scales[o];
    lut[o * 256 + m] = (h16)v;
}

// ------------- kernel 2: W codes (int32 mag + int32 sign) -> fp16 ----------
__global__ __launch_bounds__(256)
void dequant_kernel(const int* __restrict__ mag, const int* __restrict__ sgn,
                    const h16* __restrict__ lut, h16* __restrict__ Wh) {
    int o = blockIdx.x;
    const h16* lr = lut + o * 256;
    int base = o * IN_DIM + threadIdx.x * 16;
    const int4* mp = (const int4*)(mag + base);
    const int4* sp = (const int4*)(sgn + base);
    h16 t[16];
#pragma unroll
    for (int c = 0; c < 4; ++c) {
        int4 mv = mp[c];
        int4 sv = sp[c];
        h16 h0 = lr[mv.x]; t[c * 4 + 0] = (sv.x < 0) ? (h16)(-h0) : h0;
        h16 h1 = lr[mv.y]; t[c * 4 + 1] = (sv.y < 0) ? (h16)(-h1) : h1;
        h16 h2 = lr[mv.z]; t[c * 4 + 2] = (sv.z < 0) ? (h16)(-h2) : h2;
        h16 h3 = lr[mv.w]; t[c * 4 + 3] = (sv.w < 0) ? (h16)(-h3) : h3;
    }
    *(v8h*)(Wh + base)     = *(v8h*)&t[0];
    *(v8h*)(Wh + base + 8) = *(v8h*)&t[8];
}

// --------------------------- kernel 3: WMMA GEMM ---------------------------
// 256 threads = 8 waves; wave owns 64x32 patch = 4x2 wmma_f32_16x16x32_f16.
// LDS double buffer; tile k+1 streamed by async DMA while tile k is in WMMA.
__global__ __launch_bounds__(256)
void gemm_kernel(const h16* __restrict__ A, const h16* __restrict__ Wt,
                 h16* __restrict__ out) {
    __shared__ __align__(16) h16 sA[2][BM * LDK];
    __shared__ __align__(16) h16 sB[2][BN * LDK];

    const int tid   = threadIdx.x;
    const int wave  = tid >> 5;
    const int lane  = tid & 31;
    const int m0    = blockIdx.x * BM;        // x fastest -> W strip shared in L2
    const int n0    = blockIdx.y * BN;
    const int waveM = (wave & 1) * 64;
    const int waveN = (wave >> 1) * 32;

    const int lrow = lane & 15;
    // ISA per-lane K origin for 16-bit operands (CDNA5 ISA):
    const int kbA = (lane < 16) ? 0 : 8;      // A: lanes 16-31 hold K+8
    const int kbB = (lane < 16) ? 0 : 8;      // B: lanes 16-31 hold K+8

    // this thread's two 16B chunks of each tile (512 chunks total / 256 thr)
    const int r0  = tid >> 2;                 // chunk 0: c = tid
    const int ko0 = (tid & 3) * 8;
    const int r1  = (tid + 256) >> 2;         // chunk 1: c = tid + 256
    const int ko1 = ((tid + 256) & 3) * 8;

    v8f acc[4][2] = {};

#define ISSUE_TILE(buf, kk)                                                           \
    do {                                                                              \
        if (ASYNC_LDS) {                                                              \
            __builtin_amdgcn_global_load_async_to_lds_b128(                           \
                (glb_v4i_p)&A[(size_t)(m0 + r0) * IN_DIM + (kk) + ko0],               \
                (lds_v4i_p)&sA[buf][r0 * LDK + ko0], 0, 0);                           \
            __builtin_amdgcn_global_load_async_to_lds_b128(                           \
                (glb_v4i_p)&A[(size_t)(m0 + r1) * IN_DIM + (kk) + ko1],               \
                (lds_v4i_p)&sA[buf][r1 * LDK + ko1], 0, 0);                           \
            __builtin_amdgcn_global_load_async_to_lds_b128(                           \
                (glb_v4i_p)&Wt[(size_t)(n0 + r0) * IN_DIM + (kk) + ko0],              \
                (lds_v4i_p)&sB[buf][r0 * LDK + ko0], 0, 0);                           \
            __builtin_amdgcn_global_load_async_to_lds_b128(                           \
                (glb_v4i_p)&Wt[(size_t)(n0 + r1) * IN_DIM + (kk) + ko1],              \
                (lds_v4i_p)&sB[buf][r1 * LDK + ko1], 0, 0);                           \
        } else {                                                                      \
            *(v8h*)&sA[buf][r0 * LDK + ko0] =                                         \
                *(const v8h*)&A[(size_t)(m0 + r0) * IN_DIM + (kk) + ko0];             \
            *(v8h*)&sA[buf][r1 * LDK + ko1] =                                         \
                *(const v8h*)&A[(size_t)(m0 + r1) * IN_DIM + (kk) + ko1];             \
            *(v8h*)&sB[buf][r0 * LDK + ko0] =                                         \
                *(const v8h*)&Wt[(size_t)(n0 + r0) * IN_DIM + (kk) + ko0];            \
            *(v8h*)&sB[buf][r1 * LDK + ko1] =                                         \
                *(const v8h*)&Wt[(size_t)(n0 + r1) * IN_DIM + (kk) + ko1];            \
        }                                                                             \
    } while (0)

    const int nit = IN_DIM / BK;              // 128 K-steps
    ISSUE_TILE(0, 0);                         // prologue: tile 0 in flight

    for (int it = 0; it < nit; ++it) {
        const int cur = it & 1;
        if (it + 1 < nit) {
            ISSUE_TILE(cur ^ 1, (it + 1) * BK);   // stream next tile
            WAIT_ASYNC(4);                        // current tile's 4 ops done
        } else {
            WAIT_ASYNC(0);
        }
        __syncthreads();

        const h16* a_lds = sA[cur];
        const h16* b_lds = sB[cur];

        v16h af[4];
#pragma unroll
        for (int i = 0; i < 4; ++i) {
            int row = waveM + i * 16 + lrow;
            union { v16h v; v8h h[2]; } u;
            u.h[0] = *(const v8h*)&a_lds[row * LDK + kbA];       // K kb..kb+7
            u.h[1] = *(const v8h*)&a_lds[row * LDK + kbA + 16];  // K kb+16..kb+23
            af[i] = u.v;
        }
        v16h bf[2];
#pragma unroll
        for (int j = 0; j < 2; ++j) {
            int col = waveN + j * 16 + lrow;
            union { v16h v; v8h h[2]; } u;
            u.h[0] = *(const v8h*)&b_lds[col * LDK + kbB];       // K kb..kb+7
            u.h[1] = *(const v8h*)&b_lds[col * LDK + kbB + 16];  // K kb+16..kb+23
            bf[j] = u.v;
        }

#pragma unroll
        for (int i = 0; i < 4; ++i)
#pragma unroll
            for (int j = 0; j < 2; ++j)
                acc[i][j] = __builtin_amdgcn_wmma_f32_16x16x32_f16(
                    false, af[i], false, bf[j], (short)0, acc[i][j], false, false);
        __syncthreads();   // all waves done reading before buf is DMA-overwritten
    }
#undef ISSUE_TILE

    // C/D layout: VGPR r, lane l -> M = r + (l<16 ? 0 : 8), N = l%16
    const int mBase = m0 + waveM + ((lane < 16) ? 0 : 8);
    const int nBase = n0 + waveN + lrow;
#pragma unroll
    for (int i = 0; i < 4; ++i)
#pragma unroll
        for (int j = 0; j < 2; ++j)
#pragma unroll
            for (int r = 0; r < 8; ++r)
                out[(size_t)(mBase + i * 16 + r) * OUT_DIM + (nBase + j * 16)] =
                    (h16)acc[i][j][r];
}

// ---------------------------------------------------------------------------
extern "C" void kernel_launch(void* const* d_in, const int* in_sizes, int n_in,
                              void* d_out, int out_size, void* d_ws, size_t ws_size,
                              hipStream_t stream) {
    const float* x        = (const float*)d_in[0];
    const int*   mag      = (const int*)  d_in[1];
    const int*   sgn      = (const int*)  d_in[2];
    const float* scales   = (const float*)d_in[3];
    const float* alphas   = (const float*)d_in[4];
    const float* powers   = (const float*)d_in[5];
    const float* divisors = (const float*)d_in[6];
    h16* out = (h16*)d_out;   // reference output dtype is float16

    // workspace layout (~130 MB): x_fp16 | W_fp16 | LUT
    char* ws = (char*)d_ws;
    h16* xh  = (h16*)ws;                                                   // 33.6 MB
    h16* Wh  = (h16*)(ws + (size_t)M_DIM * IN_DIM * sizeof(h16));          // 90.2 MB
    h16* lut = (h16*)(ws + (size_t)M_DIM * IN_DIM * sizeof(h16)
                         + (size_t)OUT_DIM * IN_DIM * sizeof(h16));        // 5.6 MB

    cvt_x_kernel<<<(M_DIM * IN_DIM) / (8 * 256), 256, 0, stream>>>(x, xh);
    lut_kernel<<<OUT_DIM, 256, 0, stream>>>(scales, alphas, powers, divisors, lut);
    dequant_kernel<<<OUT_DIM, 256, 0, stream>>>(mag, sgn, lut, Wh);

    dim3 grid(M_DIM / BM, OUT_DIM / BN);   // M fastest: W strip + x reuse in L2
    gemm_kernel<<<grid, 256, 0, stream>>>(xh, Wh, out);
}
